// SparseResNet_3564822856157
// MI455X (gfx1250) — compile-verified, matrix operand order checked by reading, not run on hardware
//
#include <hip/hip_runtime.h>
#include <hip/hip_bf16.h>

// =====================================================================
// Sparse 3D ResNet forward for gfx1250 (MI455X).
//
// Every conv is an implicit GEMM on V_WMMA_F32_16X16X32_BF16 (f32 acc).
// Mask is binary so all bottleneck algebra folds into
//   out = [relu]( (GEMM(A,W)*gamma + beta) * mask [+ resid] ).
// Activations are NDHWC so GEMM rows are voxels, columns channels, and
// the reference weight layout (kd,kh,kw,Cin,Cout) is already [K][N].
//
// Block tile 128x64, 8 waves (4 waveM x 2 waveN), each wave 32x32 =
// 4 WMMAs per 32-K step. 1x1 convs stage A tiles with
// GLOBAL_LOAD_ASYNC_TO_LDS_B128 (ASYNCcnt path).
// =====================================================================

typedef __bf16 bhalf;
typedef bhalf v16bf __attribute__((ext_vector_type(16)));
typedef bhalf v8bf  __attribute__((ext_vector_type(8)));
typedef float v8f   __attribute__((ext_vector_type(8)));

#define TM 128
#define TN 64
#define TK 32

template <int KS, bool ASYNCA>
__global__ __launch_bounds__(256) void conv_wmma_kernel(
    const float* __restrict__ in, const float* __restrict__ wgt,
    const float* __restrict__ gamma, const float* __restrict__ beta,
    const float* __restrict__ mask, const float* __restrict__ resid,
    float* __restrict__ out,
    int Di, int doShift, int Cin, int cinShift, int Cout,
    int pad, int stride, float ascale, int doRelu)
{
    __shared__ float AsF[TM][TK + 4];   // f32 A tile; row stride 144B (16B aligned)
    __shared__ bhalf Bs[TN][TK + 8];    // bf16 B tile, [n][k]; row stride 80B

    const int tid   = threadIdx.x;
    const int lane  = tid & 31;
    const int wid   = tid >> 5;
    const int waveM = wid >> 1;        // 0..3 -> 32 rows each
    const int waveN = wid & 1;         // 0..1 -> 32 cols each

    const int m0 = blockIdx.x * TM;
    const int n0 = blockIdx.y * TN;
    const int Ktot = KS * KS * KS * Cin;

    const int dmask = (1 << doShift) - 1;
    const int vmask = (1 << (3 * doShift)) - 1;

    // ---- per-thread scalar-staging coords (KS==3 / KS==7 path) ----
    const int sRow = tid >> 1;             // 0..127
    const int sK   = (tid & 1) * 16;       // 0 or 16
    const int sm   = m0 + sRow;
    const int sb   = sm >> (3 * doShift);
    const int svox = sm & vmask;
    const int sox  = svox & dmask;
    const int soy  = (svox >> doShift) & dmask;
    const int soz  = svox >> (2 * doShift);
    const int bz   = soz * stride - pad;
    const int by   = soy * stride - pad;
    const int bx   = sox * stride - pad;

    // ---- per-thread B staging coords ----
    const int bKk = tid >> 3;              // 0..31
    const int bN  = (tid & 7) * 8;         // 0..56

    v8f accs[4] = {};

    for (int k0 = 0; k0 < Ktot; k0 += TK) {
        __syncthreads();

        // =========== stage A tile (128 x 32) ===========
        if (ASYNCA) {
            // 1x1 conv: A rows are contiguous channels in global memory.
            // Copy with async global->LDS, 16B per lane-chunk, 4 chunks/thread.
            #pragma unroll
            for (int j = 0; j < 4; ++j) {
                const int c   = tid + 256 * j;     // chunk id 0..1023
                const int row = c >> 3;            // 0..127
                const int kc  = c & 7;             // 16B chunk within 32 floats
                const int m   = m0 + row;
                const int b   = m >> (3 * doShift);
                const int vox = m & vmask;
                const int ox  = vox & dmask;
                const int oy  = (vox >> doShift) & dmask;
                const int oz  = vox >> (2 * doShift);
                const long long gidx =
                    ((((long long)b * Di + oz * stride) * Di + oy * stride) * Di +
                     ox * stride) * (long long)Cin + k0 + kc * 4;
                const unsigned long long ga =
                    (unsigned long long)(const char*)(in + gidx);
                // generic LDS pointer: low 32 bits are the LDS byte address
                const unsigned lds =
                    (unsigned)(unsigned long long)&AsF[row][kc * 4];
                asm volatile("global_load_async_to_lds_b128 %0, %1, off"
                             :: "v"(lds), "v"(ga) : "memory");
            }
            asm volatile("s_wait_asynccnt 0x0" ::: "memory");
        } else if (KS == 3) {
            // 16 consecutive k stay inside one tap (Cin >= 16): decode once.
            const int kBase = k0 + sK;
            const int tap = kBase >> cinShift;
            const int cin0 = kBase & (Cin - 1);
            const int tz = tap / 9;
            const int rr = tap - tz * 9;
            const int ty = rr / 3;
            const int tx = rr - ty * 3;
            const int iz = bz + tz, iy = by + ty, ix = bx + tx;
            const bool ok = (unsigned)iz < (unsigned)Di &&
                            (unsigned)iy < (unsigned)Di &&
                            (unsigned)ix < (unsigned)Di;
            const float* src = ok
                ? &in[((((long long)sb * Di + iz) * Di + iy) * Di + ix) *
                      (long long)Cin + cin0]
                : nullptr;
            #pragma unroll
            for (int j = 0; j < 16; ++j)
                AsF[sRow][sK + j] = src ? src[j] : 0.0f;
        } else {
            // KS==7 stem: Cin==1, per-element taps, K tail (Ktot=343).
            #pragma unroll
            for (int j = 0; j < 16; ++j) {
                const int k = k0 + sK + j;
                float v = 0.0f;
                if (k < Ktot) {
                    const int tap = k >> cinShift;
                    const int cin = k & (Cin - 1);
                    const int tz = tap / 49;
                    const int rr = tap - tz * 49;
                    const int ty = rr / 7;
                    const int tx = rr - ty * 7;
                    const int iz = bz + tz, iy = by + ty, ix = bx + tx;
                    if ((unsigned)iz < (unsigned)Di && (unsigned)iy < (unsigned)Di &&
                        (unsigned)ix < (unsigned)Di) {
                        const long long gidx =
                            ((((long long)sb * Di + iz) * Di + iy) * Di + ix) *
                            (long long)Cin + cin;
                        v = in[gidx] * ascale;
                    }
                }
                AsF[sRow][sK + j] = v;
            }
        }

        // =========== stage B tile (32 x 64) -> bf16 [n][k] ===========
        {
            const int kg = k0 + bKk;
            #pragma unroll
            for (int j = 0; j < 8; ++j) {
                float v = 0.0f;
                if (kg < Ktot) v = wgt[(long long)kg * Cout + (n0 + bN + j)];
                Bs[bN + j][bKk] = (bhalf)v;
            }
            if (kg + TK < Ktot)   // prefetch next weight tile -> global_prefetch_b8
                __builtin_prefetch(&wgt[(long long)(kg + TK) * Cout + (n0 + bN)], 0, 1);
        }
        __syncthreads();

        // =========== fragments (documented 16-bit layout) ===========
        // lanes 0-15: row=lane,    K = {kh..kh+7, 16+kh..}, kh=0
        // lanes16-31: row=lane-16, kh=8
        const int fr = lane & 15;
        const int kh = (lane >> 4) * 8;

        v16bf a0, a1, b0, b1;
        {
            const float* ar0 = &AsF[waveM * 32 + fr][0];
            const float* ar1 = &AsF[waveM * 32 + 16 + fr][0];
            #pragma unroll
            for (int i = 0; i < 8; ++i) {
                a0[i]     = (bhalf)ar0[kh + i];
                a0[i + 8] = (bhalf)ar0[16 + kh + i];
                a1[i]     = (bhalf)ar1[kh + i];
                a1[i + 8] = (bhalf)ar1[16 + kh + i];
            }
            const bhalf* br0 = &Bs[waveN * 32 + fr][0];
            const bhalf* br1 = &Bs[waveN * 32 + 16 + fr][0];
            const v8bf b0lo = *(const v8bf*)(br0 + kh);
            const v8bf b0hi = *(const v8bf*)(br0 + 16 + kh);
            const v8bf b1lo = *(const v8bf*)(br1 + kh);
            const v8bf b1hi = *(const v8bf*)(br1 + 16 + kh);
            #pragma unroll
            for (int i = 0; i < 8; ++i) {
                b0[i] = b0lo[i]; b0[i + 8] = b0hi[i];
                b1[i] = b1lo[i]; b1[i + 8] = b1hi[i];
            }
        }
        accs[0] = __builtin_amdgcn_wmma_f32_16x16x32_bf16(
            false, a0, false, b0, (short)0, accs[0], false, false);
        accs[1] = __builtin_amdgcn_wmma_f32_16x16x32_bf16(
            false, a0, false, b1, (short)0, accs[1], false, false);
        accs[2] = __builtin_amdgcn_wmma_f32_16x16x32_bf16(
            false, a1, false, b0, (short)0, accs[2], false, false);
        accs[3] = __builtin_amdgcn_wmma_f32_16x16x32_bf16(
            false, a1, false, b1, (short)0, accs[3], false, false);
    }

    // =========== epilogue: affine, mask, residual, relu ===========
    // C/D layout: lanes 0-15: N=lane, M=vgpr r; lanes 16-31: N=lane-16, M=8+r.
    const int colL = lane & 15;
    #pragma unroll
    for (int ta = 0; ta < 2; ++ta) {
        const int mrow0 = m0 + waveM * 32 + ta * 16 + ((lane >> 4) << 3);
        #pragma unroll
        for (int tb = 0; tb < 2; ++tb) {
            const int n = n0 + waveN * 32 + tb * 16 + colL;
            const float g  = gamma ? gamma[n] : 1.0f;
            const float bb = beta  ? beta[n]  : 0.0f;
            const v8f acc = accs[ta * 2 + tb];
            #pragma unroll
            for (int r = 0; r < 8; ++r) {
                const int m = mrow0 + r;
                float v = acc[r];
                v = v * g + bb;
                if (mask)  v *= mask[m];
                if (resid) v += resid[(long long)m * Cout + n];
                if (doRelu) v = fmaxf(v, 0.0f);
                out[(long long)m * Cout + n] = v;
            }
        }
    }
}

// ---------------------------------------------------------------------
// mask = (x != 0), then maxpool window 7 stride 2 SAME (64^3 -> 32^3).
// ---------------------------------------------------------------------
__global__ __launch_bounds__(256) void mask_pool7_kernel(
    const float* __restrict__ x, float* __restrict__ m, int total)
{
    int i = blockIdx.x * 256 + threadIdx.x;
    if (i >= total) return;
    int vox = i & 32767, b = i >> 15;
    int ox = vox & 31, oy = (vox >> 5) & 31, oz = vox >> 10;
    float mv = 0.0f;
    for (int tz = 0; tz < 7; ++tz) {
        int iz = oz * 2 - 2 + tz; if ((unsigned)iz >= 64u) continue;
        for (int ty = 0; ty < 7; ++ty) {
            int iy = oy * 2 - 2 + ty; if ((unsigned)iy >= 64u) continue;
            const float* row = x + (((long long)b * 64 + iz) * 64 + iy) * 64;
            for (int tx = 0; tx < 7; ++tx) {
                int ix = ox * 2 - 2 + tx; if ((unsigned)ix >= 64u) continue;
                if (row[ix] != 0.0f) mv = 1.0f;
            }
        }
    }
    m[i] = mv;
}

// maxpool window 3 stride 2 SAME on a scalar field (pad lo 0, hi 1).
__global__ __launch_bounds__(256) void mask_pool3_kernel(
    const float* __restrict__ in, float* __restrict__ out,
    int Din, int doShift, int total)
{
    int i = blockIdx.x * 256 + threadIdx.x;
    if (i >= total) return;
    int dmask = (1 << doShift) - 1;
    int b = i >> (3 * doShift);
    int vox = i & ((1 << (3 * doShift)) - 1);
    int ox = vox & dmask, oy = (vox >> doShift) & dmask, oz = vox >> (2 * doShift);
    float mv = 0.0f;
    for (int tz = 0; tz < 3; ++tz) { int iz = 2 * oz + tz; if (iz >= Din) break;
        for (int ty = 0; ty < 3; ++ty) { int iy = 2 * oy + ty; if (iy >= Din) break;
            for (int tx = 0; tx < 3; ++tx) { int ix = 2 * ox + tx; if (ix >= Din) break;
                mv = fmaxf(mv, in[(((long long)b * Din + iz) * Din + iy) * Din + ix]);
            } } }
    out[i] = mv;
}

// maxpool window 3 stride 2 SAME, channels-last activation.
__global__ __launch_bounds__(256) void maxpool3c_kernel(
    const float* __restrict__ in, float* __restrict__ out,
    int Din, int doShift, int cShift, long long total)
{
    long long i = (long long)blockIdx.x * 256 + threadIdx.x;
    if (i >= total) return;
    int C = 1 << cShift;
    int c = (int)(i & (C - 1));
    long long v = i >> cShift;
    int dmask = (1 << doShift) - 1;
    int b = (int)(v >> (3 * doShift));
    int vox = (int)(v & ((1 << (3 * doShift)) - 1));
    int ox = vox & dmask, oy = (vox >> doShift) & dmask, oz = vox >> (2 * doShift);
    float val = -__builtin_inff();
    for (int tz = 0; tz < 3; ++tz) { int iz = 2 * oz + tz; if (iz >= Din) break;
        for (int ty = 0; ty < 3; ++ty) { int iy = 2 * oy + ty; if (iy >= Din) break;
            for (int tx = 0; tx < 3; ++tx) { int ix = 2 * ox + tx; if (ix >= Din) break;
                long long gi = (((((long long)b * Din + iz) * Din + iy) * Din + ix) << cShift) | c;
                val = fmaxf(val, in[gi]);
            } } }
    out[i] = val;
}

// m[::2,::2,::2] strided subsample.
__global__ __launch_bounds__(256) void subsample2_kernel(
    const float* __restrict__ in, float* __restrict__ out,
    int Din, int doShift, int total)
{
    int i = blockIdx.x * 256 + threadIdx.x;
    if (i >= total) return;
    int dmask = (1 << doShift) - 1;
    int b = i >> (3 * doShift);
    int vox = i & ((1 << (3 * doShift)) - 1);
    int ox = vox & dmask, oy = (vox >> doShift) & dmask, oz = vox >> (2 * doShift);
    out[i] = in[(((long long)b * Din + 2 * oz) * Din + 2 * oy) * Din + 2 * ox];
}

// [B, 4,4,4, 1024] NDHWC -> [B, 1024, 4,4,4] NCDHW.
__global__ __launch_bounds__(256) void nchw_transpose_kernel(
    const float* __restrict__ in, float* __restrict__ out, int total)
{
    int i = blockIdx.x * 256 + threadIdx.x;
    if (i >= total) return;
    int vox = i & 63;
    int c   = (i >> 6) & 1023;
    int b   = i >> 16;
    out[i] = in[(((long long)(b << 6) | vox) << 10) + c];
}

// =====================================================================
// Host orchestration
// =====================================================================
static int ilog2i(int v) { int s = 0; while ((1 << s) < v) ++s; return s; }

static void launch_conv(hipStream_t st, const float* in, const float* w,
                        const float* g, const float* b, const float* mask,
                        const float* resid, float* out,
                        int Di, int Do, int Cin, int Cout, int ks, int pad,
                        int stride, float ascale, int relu)
{
    const long long M = 16LL * Do * Do * Do;
    dim3 grid((unsigned)(M / TM), (unsigned)(Cout / TN), 1);
    const int doS = ilog2i(Do), ciS = ilog2i(Cin);
    if (ks == 1) {
        conv_wmma_kernel<1, true><<<grid, dim3(256), 0, st>>>(
            in, w, g, b, mask, resid, out, Di, doS, Cin, ciS, Cout,
            pad, stride, ascale, relu);
    } else if (ks == 3) {
        conv_wmma_kernel<3, false><<<grid, dim3(256), 0, st>>>(
            in, w, g, b, mask, resid, out, Di, doS, Cin, ciS, Cout,
            pad, stride, ascale, relu);
    } else {
        conv_wmma_kernel<7, false><<<grid, dim3(256), 0, st>>>(
            in, w, g, b, mask, resid, out, Di, doS, Cin, ciS, Cout,
            pad, stride, ascale, relu);
    }
}

struct BlockP {
    const float *w1, *g1, *b1, *w2, *g2, *b2, *w3, *g3, *b3, *wd, *gd, *bd;
};

extern "C" void kernel_launch(void* const* d_in, const int* in_sizes, int n_in,
                              void* d_out, int out_size, void* d_ws, size_t ws_size,
                              hipStream_t stream)
{
    (void)in_sizes; (void)n_in; (void)out_size; (void)ws_size;

    // ---- inputs in setup_inputs() dict (insertion) order ----
    int ii = 0;
    const float* x      = (const float*)d_in[ii++];
    const float* stem_w = (const float*)d_in[ii++];
    const float* stem_g = (const float*)d_in[ii++];
    const float* stem_b = (const float*)d_in[ii++];

    BlockP blocks[13];
    const int counts[3] = {3, 4, 6};
    int nb = 0;
    for (int s = 0; s < 3; ++s) {
        for (int j = 0; j < counts[s]; ++j) {
            BlockP& p = blocks[nb++];
            p.w1 = (const float*)d_in[ii++]; p.g1 = (const float*)d_in[ii++]; p.b1 = (const float*)d_in[ii++];
            p.w2 = (const float*)d_in[ii++]; p.g2 = (const float*)d_in[ii++]; p.b2 = (const float*)d_in[ii++];
            p.w3 = (const float*)d_in[ii++]; p.g3 = (const float*)d_in[ii++]; p.b3 = (const float*)d_in[ii++];
            if (j == 0) {
                p.wd = (const float*)d_in[ii++]; p.gd = (const float*)d_in[ii++]; p.bd = (const float*)d_in[ii++];
            } else { p.wd = p.gd = p.bd = nullptr; }
        }
    }

    // ---- workspace partition (B=16) ----
    char* ws = (char*)d_ws;
    size_t off = 0;
    auto alloc = [&](size_t bytes) -> float* {
        float* p = (float*)(ws + off);
        off += (bytes + 255) & ~(size_t)255;
        return p;
    };
    float* m32 = alloc(16LL * 32768 * 4);          //  2 MB  mask @32^3
    float* m16 = alloc(16LL * 4096 * 4);           //  mask @16^3
    float* m8  = alloc(16LL * 512 * 4);            //  mask @8^3
    float* m4  = alloc(16LL * 64 * 4);             //  mask @4^3
    float* P0  = alloc(16LL * 32768 * 64 * 4);     // 134 MB (stem out / ping)
    float* P1  = alloc(16LL * 4096 * 256 * 4);     //  67 MB (pong)
    float* Q   = alloc(16LL * 4096 * 64 * 4);      //  17 MB (pooled stem)
    float* T1  = alloc(16LL * 4096 * 64 * 4);      //  17 MB (inner act)
    float* T2  = alloc(16LL * 4096 * 64 * 4);      //  17 MB (inner act)
    float* R   = alloc(16LL * 4096 * 256 * 4);     //  67 MB (downsample resid)

    // ---- masks ----
    mask_pool7_kernel<<<(16 * 32768 + 255) / 256, 256, 0, stream>>>(x, m32, 16 * 32768);
    mask_pool3_kernel<<<(16 * 4096 + 255) / 256, 256, 0, stream>>>(m32, m16, 32, 4, 16 * 4096);
    subsample2_kernel<<<(16 * 512 + 255) / 256, 256, 0, stream>>>(m16, m8, 16, 3, 16 * 512);
    subsample2_kernel<<<(16 * 64 + 255) / 256, 256, 0, stream>>>(m8, m4, 8, 2, 16 * 64);

    // ---- stem: conv7^3 s2 of (x*100), affine, *m32, relu -> P0 (32^3 x 64) ----
    launch_conv(stream, x, stem_w, stem_g, stem_b, m32, nullptr, P0,
                64, 32, 1, 64, 7, 2, 2, 100.0f, 1);
    // maxpool 3 s2 -> Q (16^3 x 64)
    maxpool3c_kernel<<<(unsigned)((16LL * 4096 * 64 + 255) / 256), 256, 0, stream>>>(
        P0, Q, 32, 4, 6, 16LL * 4096 * 64);

    // ---- residual stages ----
    const float* stageMask[3] = {m16, m8, m4};
    const int stageD[3]  = {16, 8, 4};
    const int innerC[3]  = {64, 128, 256};
    const int outC[3]    = {256, 512, 1024};

    const float* cur = Q;
    int curC = 64;
    float* ping = P1;
    float* pong = P0;
    nb = 0;
    for (int s = 0; s < 3; ++s) {
        for (int j = 0; j < counts[s]; ++j) {
            const BlockP& p = blocks[nb++];
            const int stride = (s > 0 && j == 0) ? 2 : 1;
            const int Do  = stageD[s];
            const int Din = (stride == 2) ? Do * 2 : Do;
            const float* mo = stageMask[s];
            const int ic = innerC[s], oc = outC[s];

            // o1 = relu((conv1x1_s(x)*g1 + b1) * m)
            launch_conv(stream, cur, p.w1, p.g1, p.b1, mo, nullptr, T1,
                        Din, Do, curC, ic, 1, 0, stride, 1.0f, 1);
            // o2 = relu((conv3x3(o1)*g2 + b2) * m)
            launch_conv(stream, T1, p.w2, p.g2, p.b2, mo, nullptr, T2,
                        Do, Do, ic, ic, 3, 1, 1, 1.0f, 1);
            // residual
            const float* rptr;
            if (p.wd) {   // r = (conv1x1_s(x)*gd + bd) * m
                launch_conv(stream, cur, p.wd, p.gd, p.bd, mo, nullptr, R,
                            Din, Do, curC, oc, 1, 0, stride, 1.0f, 0);
                rptr = R;
            } else {
                rptr = cur;   // identity (already masked, same shape)
            }
            // out = relu((conv1x1(o2)*g3 + b3) * m + r)
            float* dst = ping;
            launch_conv(stream, T2, p.w3, p.g3, p.b3, mo, rptr, dst,
                        Do, Do, ic, oc, 1, 0, 1, 1.0f, 1);

            cur = dst; curC = oc;
            float* t = ping; ping = pong; pong = t;
        }
    }

    // ---- NDHWC (B,4,4,4,1024) -> NCDHW d_out ----
    nchw_transpose_kernel<<<(16 * 65536 + 255) / 256, 256, 0, stream>>>(
        cur, (float*)d_out, 16 * 65536);
}
